// SparseMultiHeadAttention_45303315038789
// MI455X (gfx1250) — matrix operand order repaired and checked by reading, not compiled, not run on hardware
//
#include <hip/hip_runtime.h>

typedef __attribute__((ext_vector_type(16))) __bf16 v16bf;
typedef __attribute__((ext_vector_type(8)))  float  v8f;
typedef __attribute__((ext_vector_type(4)))  unsigned int v4u;
typedef __attribute__((ext_vector_type(8)))  int    v8i;
typedef __attribute__((ext_vector_type(4)))  int    v4i;

#define N_TOK 4096
#define C_DIM 1024
#define H_NUM 16
#define HD    64
#define FW    2    // waves per block in flash kernel

#if defined(__has_builtin)
#  if __has_builtin(__builtin_amdgcn_tensor_load_to_lds) && __has_builtin(__builtin_amdgcn_s_wait_tensorcnt)
#    define HAVE_TDM 1
#  else
#    define HAVE_TDM 0
#  endif
#else
#  define HAVE_TDM 0
#endif

#define LDS_FENCE() asm volatile("" ::: "memory")

#if HAVE_TDM
// Issue one TDM copy: 2-D tile (tile_d0 x tile_d1) of 8-byte units, row stride stride0 units.
static __device__ __forceinline__ void tdm_load(unsigned long long gaddr, unsigned lds_off,
                                                unsigned tile_d0, unsigned tile_d1,
                                                unsigned tensor_d0, unsigned tensor_d1,
                                                unsigned long long stride0) {
  v4u g0; v8i g1; v4i gz = {};
  g0[0] = 1u;                                        // count=1, no gather
  g0[1] = lds_off;                                   // LDS byte address
  g0[2] = (unsigned)gaddr;                           // global addr [31:0]
  g0[3] = (unsigned)(gaddr >> 32) | (2u << 30);      // global addr [56:32] | type=2
  g1[0] = (int)(3u << 16);                           // workgroup_mask=0, data_size=8B
  g1[1] = (int)((tensor_d0 & 0xffffu) << 16);        // tensor_dim0[15:0]
  g1[2] = (int)(((tensor_d0 >> 16) & 0xffffu) | ((tensor_d1 & 0xffffu) << 16));
  g1[3] = (int)(((tensor_d1 >> 16) & 0xffffu) | ((tile_d0 & 0xffffu) << 16));
  g1[4] = (int)(tile_d1 & 0xffffu);                  // tile_dim1 | tile_dim2=0
  g1[5] = (int)(unsigned)stride0;                    // dim0 stride [31:0]
  g1[6] = (int)((unsigned)(stride0 >> 32) & 0xffffu);// dim0 stride [47:32] | dim1 stride lo=0
  g1[7] = 0;
#if __clang_major__ >= 23
  v8i gz8 = {};
  __builtin_amdgcn_tensor_load_to_lds(g0, g1, gz, gz, gz8, 0);
#else
  __builtin_amdgcn_tensor_load_to_lds(g0, g1, gz, gz, 0);
#endif
}
#endif

static __device__ __forceinline__ float wave_allreduce_sum(float v) {
#pragma unroll
  for (int m = 16; m >= 1; m >>= 1) v += __shfl_xor(v, m, 32);
  return v;
}

static __device__ __forceinline__ v16bf load_a_frag(const __bf16* __restrict__ p) {
  // A layout: lane<16 -> K{0..7,16..23}; lane>=16 -> K{8..15,24..31} (hlf*8 pre-added by caller)
  v16bf a;
#pragma unroll
  for (int i = 0; i < 8; ++i) { a[i] = p[i]; a[8 + i] = p[16 + i]; }
  return a;
}

__global__ __launch_bounds__(256) void cvt_f32_to_bf16(const float* __restrict__ src,
                                                       __bf16* __restrict__ dst, int n) {
  int i = blockIdx.x * blockDim.x + threadIdx.x;
  if (i < n) dst[i] = (__bf16)src[i];
}

// C[M][Nc] (f32) = A[M][K] (bf16 row-major) @ B[Nc][K]^T (bf16 weights [out][in]).
// Block: 8 waves share one 64-col N tile; B staged in LDS (64x64 bf16 = 8KB) via
// double-buffered TDM; K stepped by 64. All 8 B fragments loaded up-front so the
// 8 WMMAs run back-to-back behind a single DScnt wait.
__global__ __launch_bounds__(256) void gemm_wmma_bf16(const __bf16* __restrict__ A,
                                                      const __bf16* __restrict__ B,
                                                      float* __restrict__ C,
                                                      int M, int Nc, int K, int tilesN) {
  __shared__ __bf16 Bl[2][64][64];   // [buf][n_local][k_local]
  int w = threadIdx.x >> 5;
  int bm = blockIdx.x / tilesN, bn = blockIdx.x % tilesN;
  int mbase = bm * 128 + w * 16, nbase = bn * 64;
  int lane = threadIdx.x & 31;
  int hlf = lane >> 4, lrow = lane & 15;
  bool issuer = (threadIdx.x < 32);

  const __bf16* arow = A + (size_t)(mbase + lrow) * K + hlf * 8;
  v8f acc[4] = {};

#if HAVE_TDM
  unsigned long long bbase = (unsigned long long)(size_t)(B + (size_t)nbase * K);
  unsigned ld0 = (unsigned)(size_t)&Bl[0][0][0];
  unsigned ld1 = (unsigned)(size_t)&Bl[1][0][0];
  unsigned sK8 = (unsigned)K / 4;                     // row stride / width in 8B units
  if (issuer) tdm_load(bbase, ld0, 16, 64, sK8, 64, sK8);   // 64 rows x 128B
#endif
  v16bf a_cur[2], a_nxt[2];
  a_cur[0] = load_a_frag(arow);
  a_cur[1] = load_a_frag(arow + 32);
  a_nxt[0] = a_cur[0]; a_nxt[1] = a_cur[1];
  int cur = 0;

  for (int kb = 0; kb < K; kb += 64) {
    bool hn = (kb + 64) < K;
    if (hn) {
      a_nxt[0] = load_a_frag(arow + kb + 64);
      a_nxt[1] = load_a_frag(arow + kb + 96);
    }
#if HAVE_TDM
    if (issuer) {
      if (hn) {
        tdm_load(bbase + (unsigned long long)(kb + 64) * 2,
                 cur ? ld0 : ld1, 16, 64, sK8, 64, sK8);
        __builtin_amdgcn_s_wait_tensorcnt(1);
      } else {
        __builtin_amdgcn_s_wait_tensorcnt(0);
      }
    }
    LDS_FENCE();
    __syncthreads();
#else
    for (int idx = threadIdx.x; idx < 64 * 64; idx += 256)
      Bl[cur][idx >> 6][idx & 63] = B[(size_t)(nbase + (idx >> 6)) * K + kb + (idx & 63)];
    __syncthreads();
#endif
    {
      v16bf bt[2][4];  // all eight B fragments first -> one wait, 8 chained WMMAs
#pragma unroll
      for (int kc = 0; kc < 2; ++kc)
#pragma unroll
        for (int t = 0; t < 4; ++t)
#pragma unroll
          for (int i = 0; i < 16; ++i)
            bt[kc][t][i] = Bl[cur][t * 16 + lrow][kc * 32 + hlf * 8 + (i >> 3) * 16 + (i & 7)];
#pragma unroll
      for (int kc = 0; kc < 2; ++kc)
#pragma unroll
        for (int t = 0; t < 4; ++t)
          acc[t] = __builtin_amdgcn_wmma_f32_16x16x32_bf16(false, a_cur[kc], false, bt[kc][t],
                                                           (short)0, acc[t], false, false);
    }
    __syncthreads();
    a_cur[0] = a_nxt[0]; a_cur[1] = a_nxt[1];
    cur ^= 1;
  }
#pragma unroll
  for (int t = 0; t < 4; ++t)
#pragma unroll
    for (int g = 0; g < 8; ++g)
      C[(size_t)(mbase + hlf * 8 + g) * Nc + nbase + t * 16 + lrow] = acc[t][g];
}

// One wave per (n, h): RMS-norm q/k + RoPE. Outputs:
//   q, k : [H][N_TOK][HD] bf16 (q pre-scaled by 1/sqrt(HD))
//   v    : [H][N_TOK/32][HD][32] bf16 (32-key block-transposed, TDM/fragment friendly)
__global__ __launch_bounds__(256) void rms_rope_split(const float* __restrict__ qkv,
                                                      const int* __restrict__ coords,
                                                      const float* __restrict__ gq,
                                                      const float* __restrict__ gk,
                                                      __bf16* __restrict__ qo,
                                                      __bf16* __restrict__ ko,
                                                      __bf16* __restrict__ vo) {
  int wid  = blockIdx.x * (blockDim.x >> 5) + (threadIdx.x >> 5);
  int n = wid >> 4, h = wid & 15;
  int lane = threadIdx.x & 31;
  const float* base = qkv + (size_t)n * 3 * C_DIM + h * HD;

  float q0 = base[2 * lane],             q1 = base[2 * lane + 1];
  float k0 = base[C_DIM + 2 * lane],     k1 = base[C_DIM + 2 * lane + 1];
  float v0 = base[2 * C_DIM + 2 * lane], v1 = base[2 * C_DIM + 2 * lane + 1];

  float ssq = wave_allreduce_sum(q0 * q0 + q1 * q1);
  float ssk = wave_allreduce_sum(k0 * k0 + k1 * k1);
  float invq = 8.0f / fmaxf(sqrtf(ssq), 1e-12f);
  float invk = 8.0f / fmaxf(sqrtf(ssk), 1e-12f);

  float gq0 = gq[h * HD + 2 * lane], gq1 = gq[h * HD + 2 * lane + 1];
  float gk0 = gk[h * HD + 2 * lane], gk1 = gk[h * HD + 2 * lane + 1];
  q0 *= invq * gq0; q1 *= invq * gq1;
  k0 *= invk * gk0; k1 *= invk * gk1;

  int p = lane >> 3, f = lane & 7;
  float pos  = (float)coords[n * 5 + 1 + p];
  float freq = __expf(-(float)f * 0.14391565f);   // ln(10000)/64
  float sn, cs;
  __sincosf(pos * freq, &sn, &cs);

  float qre = q0 * cs - q1 * sn, qim = q0 * sn + q1 * cs;
  float kre = k0 * cs - k1 * sn, kim = k0 * sn + k1 * cs;

  size_t o = ((size_t)h * N_TOK + n) * HD + 2 * lane;
  qo[o] = (__bf16)(qre * 0.125f); qo[o + 1] = (__bf16)(qim * 0.125f);
  ko[o] = (__bf16)kre;            ko[o + 1] = (__bf16)kim;

  size_t ob = (((size_t)h * (N_TOK / 32) + (n >> 5)) * HD + 2 * lane) * 32 + (n & 31);
  vo[ob] = (__bf16)v0; vo[ob + 32] = (__bf16)v1;
}

// Flash attention: one wave per (head, 16-row q block); 32 keys/iter.
// K tile (32x64) and V tile (64x32, pre-transposed) are contiguous 4KB blocks,
// staged via double-buffered TDM. All LDS exchanges are wave-local (same-wave
// LDS ops are in-order per ISA), so no block barriers in the main loop.
__global__ __launch_bounds__(64) void flash_attn_wmma(const __bf16* __restrict__ q,
                                                      const __bf16* __restrict__ k,
                                                      const __bf16* __restrict__ v,
                                                      __bf16* __restrict__ attn) {
  __shared__ __bf16 KT[FW][2][32 * 64];   // [wave][buf][key*64 + ch]
  __shared__ __bf16 VT[FW][2][64 * 32];   // [wave][buf][ch*32 + key]
  __shared__ float  S [FW][16][32];
  __shared__ __bf16 P [FW][16][32];
  __shared__ float  CR[FW][16];

  int w    = threadIdx.x >> 5;
  int wid  = blockIdx.x * FW + w;
  int h    = wid >> 8;
  int qbase = (wid & 255) * 16;
  int lane = threadIdx.x & 31;
  int hlf  = lane >> 4, lrow = lane & 15;

  const __bf16* qrow = q + ((size_t)h * N_TOK + qbase + lrow) * HD;
  v16bf qa[2];
#pragma unroll
  for (int kc = 0; kc < 2; ++kc)
#pragma unroll
    for (int i = 0; i < 8; ++i) {
      qa[kc][i]     = qrow[kc * 32 + hlf * 8 + i];
      qa[kc][8 + i] = qrow[kc * 32 + 16 + hlf * 8 + i];
    }

  v8f acc[4] = {};
  float m_i = -1e30f, l_i = 0.0f;

  const __bf16* kh = k + (size_t)h * N_TOK * HD;
  const __bf16* vh = v + (size_t)h * (N_TOK / 32) * HD * 32;

#if HAVE_TDM
  unsigned ldK0 = (unsigned)(size_t)&KT[w][0][0], ldK1 = (unsigned)(size_t)&KT[w][1][0];
  unsigned ldV0 = (unsigned)(size_t)&VT[w][0][0], ldV1 = (unsigned)(size_t)&VT[w][1][0];
  unsigned long long kg = (unsigned long long)(size_t)kh;
  unsigned long long vg = (unsigned long long)(size_t)vh;
  // contiguous 4KB = 512 x 8B units, 1-D copy
  tdm_load(kg, ldK0, 512, 1, 512, 1, 512);
  tdm_load(vg, ldV0, 512, 1, 512, 1, 512);
#endif

  for (int jb = 0; jb < N_TOK; jb += 32) {
    int cur = (jb >> 5) & 1;
#if HAVE_TDM
    if (jb + 32 < N_TOK) {
      tdm_load(kg + (unsigned long long)(jb + 32) * HD * 2, cur ? ldK0 : ldK1, 512, 1, 512, 1, 512);
      tdm_load(vg + (unsigned long long)((jb >> 5) + 1) * HD * 32 * 2, cur ? ldV0 : ldV1, 512, 1, 512, 1, 512);
      __builtin_amdgcn_s_wait_tensorcnt(2);
    } else {
      __builtin_amdgcn_s_wait_tensorcnt(0);
    }
    LDS_FENCE();
#else
    cur = 0;
    {
      const __bf16* kp = kh + (size_t)jb * HD;
      const __bf16* vp = vh + (size_t)(jb >> 5) * HD * 32;
      for (int idx = lane; idx < 32 * 64; idx += 32) {
        KT[w][0][idx] = kp[idx];
        VT[w][0][idx] = vp[idx];
      }
      LDS_FENCE();
    }
#endif
    // S = Q @ K^T : load all four B fragments, then chain four WMMAs
    {
      v16bf kbf[2][2];   // [key-tile][k-chunk]
#pragma unroll
      for (int t = 0; t < 2; ++t)
#pragma unroll
        for (int kc = 0; kc < 2; ++kc)
#pragma unroll
          for (int i = 0; i < 16; ++i)
            kbf[t][kc][i] = KT[w][cur][(t * 16 + lrow) * 64 + kc * 32 + hlf * 8 + (i >> 3) * 16 + (i & 7)];
#pragma unroll
      for (int t = 0; t < 2; ++t) {
        v8f s = {};
        s = __builtin_amdgcn_wmma_f32_16x16x32_bf16(false, qa[0], false, kbf[t][0],
                                                    (short)0, s, false, false);
        s = __builtin_amdgcn_wmma_f32_16x16x32_bf16(false, qa[1], false, kbf[t][1],
                                                    (short)0, s, false, false);
#pragma unroll
        for (int g = 0; g < 8; ++g)
          S[w][hlf * 8 + g][t * 16 + lrow] = s[g];
      }
    }
    LDS_FENCE();

    // online softmax: lane r (<16) owns row r
    if (lane < 16) {
      float mx = m_i;
#pragma unroll 8
      for (int j = 0; j < 32; ++j) mx = fmaxf(mx, S[w][lane][j]);
      float corr = __expf(m_i - mx);
      float ps = 0.0f;
#pragma unroll 8
      for (int j = 0; j < 32; ++j) {
        float pv = __expf(S[w][lane][j] - mx);
        ps += pv;
        P[w][lane][j] = (__bf16)pv;
      }
      m_i = mx;
      l_i = l_i * corr + ps;
      CR[w][lane] = corr;
    }
    LDS_FENCE();

    // P @ V : load P fragment, V fragments and correction factors, then chain WMMAs
    {
      v16bf pa;
#pragma unroll
      for (int i = 0; i < 8; ++i) {
        pa[i]     = P[w][lrow][hlf * 8 + i];
        pa[8 + i] = P[w][lrow][16 + hlf * 8 + i];
      }
      v16bf vbf[4];
#pragma unroll
      for (int t = 0; t < 4; ++t)
#pragma unroll
        for (int i = 0; i < 16; ++i)
          vbf[t][i] = VT[w][cur][(t * 16 + lrow) * 32 + hlf * 8 + (i >> 3) * 16 + (i & 7)];
      float cg[8];
#pragma unroll
      for (int g = 0; g < 8; ++g) cg[g] = CR[w][hlf * 8 + g];
#pragma unroll
      for (int t = 0; t < 4; ++t) {
#pragma unroll
        for (int g = 0; g < 8; ++g) acc[t][g] *= cg[g];
        acc[t] = __builtin_amdgcn_wmma_f32_16x16x32_bf16(false, pa, false, vbf[t],
                                                         (short)0, acc[t], false, false);
      }
    }
    LDS_FENCE();
  }

  if (lane < 16) CR[w][lane] = l_i;
  LDS_FENCE();
  {
    float rs[8];
#pragma unroll
    for (int g = 0; g < 8; ++g) rs[g] = CR[w][hlf * 8 + g];
#pragma unroll
    for (int t = 0; t < 4; ++t)
#pragma unroll
      for (int g = 0; g < 8; ++g) {
        int row = hlf * 8 + g, col = t * 16 + lrow;
        attn[(size_t)(qbase + row) * C_DIM + h * HD + col] = (__bf16)(acc[t][g] / rs[g]);
      }
  }
}

extern "C" void kernel_launch(void* const* d_in, const int* in_sizes, int n_in,
                              void* d_out, int out_size, void* d_ws, size_t ws_size,
                              hipStream_t stream) {
  (void)in_sizes; (void)n_in; (void)out_size; (void)ws_size;
  const float* x      = (const float*)d_in[0];
  const int*   coords = (const int*)d_in[1];
  const float* wqkv   = (const float*)d_in[2];
  const float* wout   = (const float*)d_in[3];
  const float* gq     = (const float*)d_in[4];
  const float* gk     = (const float*)d_in[5];
  float* out = (float*)d_out;

  char* p = (char*)d_ws;
  __bf16* xb    = (__bf16*)p; p += (size_t)N_TOK * C_DIM * 2;
  __bf16* wqkvb = (__bf16*)p; p += (size_t)3 * C_DIM * C_DIM * 2;
  __bf16* woutb = (__bf16*)p; p += (size_t)C_DIM * C_DIM * 2;
  __bf16* qb    = (__bf16*)p; p += (size_t)N_TOK * C_DIM * 2;
  __bf16* kb    = (__bf16*)p; p += (size_t)N_TOK * C_DIM * 2;
  __bf16* vb    = (__bf16*)p; p += (size_t)N_TOK * C_DIM * 2;
  __bf16* attnb = (__bf16*)p; p += (size_t)N_TOK * C_DIM * 2;
  float*  qkv   = (float*)p;

  int nx = N_TOK * C_DIM, nwq = 3 * C_DIM * C_DIM, nwo = C_DIM * C_DIM;
  cvt_f32_to_bf16<<<(nx + 255) / 256, 256, 0, stream>>>(x, xb, nx);
  cvt_f32_to_bf16<<<(nwq + 255) / 256, 256, 0, stream>>>(wqkv, wqkvb, nwq);
  cvt_f32_to_bf16<<<(nwo + 255) / 256, 256, 0, stream>>>(wout, woutb, nwo);

  // qkv = x @ w_qkv^T : (4096/128)=32 M-blocks x (3072/64)=48 N-tiles
  gemm_wmma_bf16<<<32 * 48, 256, 0, stream>>>(xb, wqkvb, qkv, N_TOK, 3 * C_DIM, C_DIM, 48);

  rms_rope_split<<<(N_TOK * H_NUM) / 8, 256, 0, stream>>>(qkv, coords, gq, gk, qb, kb, vb);

  flash_attn_wmma<<<(H_NUM * (N_TOK / 16)) / FW, 32 * FW, 0, stream>>>(qb, kb, vb, attnb);

  // out = attn @ w_out^T : 32 M-blocks x 16 N-tiles
  gemm_wmma_bf16<<<32 * 16, 256, 0, stream>>>(attnb, woutb, out, N_TOK, C_DIM, C_DIM, 16);
}
